// SupConLoss_85907935854913
// MI455X (gfx1250) — matrix operand order repaired and checked by reading, not compiled, not run on hardware
//
#include <hip/hip_runtime.h>
#include <hip/hip_fp16.h>

typedef __attribute__((ext_vector_type(16))) _Float16 v16h;
typedef __attribute__((ext_vector_type(8)))  float    v8f;

#define TEMP_INV 14.2857142857142857f   // 1/0.07
#define EPSV     1e-12f
#define DIMK     512                    // feature dim (known from reference)
#define NT       32                     // columns per LDS tile
#define KC       (DIMK / 32)            // 16 K-chunks of 32

// ---------------- Kernel 1: L2-normalize rows, emit f16 matrix ----------------
__global__ __launch_bounds__(256)
void normalize_f16(const float* __restrict__ f, _Float16* __restrict__ h, int Dv) {
    int row  = blockIdx.x * 8 + (threadIdx.x >> 5);
    int lane = threadIdx.x & 31;
    const float* fr = f + (size_t)row * Dv;
    float ss = 0.f;
    for (int k = lane; k < Dv; k += 32) { float v = fr[k]; ss += v * v; }
    #pragma unroll
    for (int m = 16; m >= 1; m >>= 1) ss += __shfl_xor(ss, m, 32);
    float inv = rsqrtf(ss);
    _Float16* hr = h + (size_t)row * Dv;
    for (int k = lane; k < Dv; k += 32) hr[k] = (_Float16)(fr[k] * inv);
}

// Issue one tile (NT cols x DIMK halves = 32 KB) of async global->LDS copies.
// 2048 16-byte chunks over 256 threads -> 8 async b128 issues per thread.
__device__ __forceinline__ void stage_tile_async(const _Float16* __restrict__ h,
                                                 int ct, _Float16* dst, int tid) {
    #pragma unroll
    for (int j = 0; j < 8; ++j) {
        int q = tid + 256 * j;
        unsigned ldsOff = (unsigned)(uintptr_t)(dst + q * 8);       // LDS byte offset (addr[31:0])
        const _Float16* src = h + (size_t)ct * DIMK + q * 8;
        asm volatile("global_load_async_to_lds_b128 %0, %1, off"
                     :: "v"(ldsOff), "v"(src) : "memory");
    }
}

// ---------------- Kernel 2: fused sim-GEMM + online softmax + masks ----------------
// Block = 256 threads = 8 waves; wave w owns rows [blk*128 + 16w, +16).
// Double-buffered 32-column B tiles staged via GLOBAL_LOAD_ASYNC_TO_LDS_B128,
// overlapped with WMMA compute on the current tile.
__global__ __launch_bounds__(256)
void supcon_main(const _Float16* __restrict__ h, const int* __restrict__ lab,
                 float* __restrict__ contrib, float* __restrict__ validOut, int B) {
    __shared__ _Float16 Bt[2][NT * DIMK];   // 2 x 32 KB

    const int tid  = threadIdx.x;
    const int wave = tid >> 5;
    const int lane = tid & 31;
    const int l16  = lane & 15;
    const bool hi  = lane >= 16;
    const int rowBase = blockIdx.x * 128 + wave * 16;

    // Prologue: kick off tile 0 before doing register setup (overlap).
    stage_tile_async(h, 0, Bt[0], tid);

    // ---- Resident A fragments: 16 rows x 512 K of f16 (contiguous k = 16h + s order) ----
    // lanes 0-15 hold K = kc+0..15 ; lanes 16-31 hold kc+16..31
    const int ar   = rowBase + l16;
    const int aoff = hi ? 16 : 0;
    v16h aF[KC];
    #pragma unroll
    for (int c = 0; c < KC; ++c) {
        union { v16h v; uint4 q[2]; } u;
        const _Float16* p = h + (size_t)ar * DIMK + 32 * c + aoff;
        u.q[0] = *(const uint4*)(p);
        u.q[1] = *(const uint4*)(p + 8);
        aF[c] = u.v;
    }

    // Per-lane state for the 8 C-matrix rows of this lane's half-group.
    int   rowLab[8], rowGr[8];
    float runM[8], runS[8], posS[8], posC[8];
    #pragma unroll
    for (int i = 0; i < 8; ++i) {
        int gr = rowBase + i + (hi ? 8 : 0);
        rowGr[i]  = gr;
        rowLab[i] = lab[gr];
        runM[i] = -3.0e38f; runS[i] = 0.f; posS[i] = 0.f; posC[i] = 0.f;
    }

    const int boff = hi ? 16 : 0;   // same contiguous k order as A: high lanes take K+16..31
    const int nTiles = B / NT;

    for (int t = 0; t < nTiles; ++t) {
        const int ct = t * NT;
        // Current tile's async copies complete (in-order), then block-wide sync.
        asm volatile("s_wait_asynccnt 0x0" ::: "memory");
        __syncthreads();
        // Start DMA of the next tile into the other buffer; overlaps compute.
        if (t + 1 < nTiles) stage_tile_async(h, ct + NT, Bt[(t + 1) & 1], tid);

        const _Float16* Bcur = Bt[t & 1];
        v8f acc0 = {}; v8f acc1 = {};
        #pragma unroll
        for (int c = 0; c < KC; ++c) {
            union { v16h v; uint4 q[2]; } b0, b1;
            const _Float16* q0 = Bcur + (size_t)l16 * DIMK + 32 * c + boff;
            const _Float16* q1 = Bcur + (size_t)(16 + l16) * DIMK + 32 * c + boff;
            b0.q[0] = *(const uint4*)(q0); b0.q[1] = *(const uint4*)(q0 + 8);
            b1.q[0] = *(const uint4*)(q1); b1.q[1] = *(const uint4*)(q1 + 8);
            acc0 = __builtin_amdgcn_wmma_f32_16x16x32_f16(
                false, aF[c], false, b0.v, (short)0, acc0, false, false);
            acc1 = __builtin_amdgcn_wmma_f32_16x16x32_f16(
                false, aF[c], false, b1.v, (short)0, acc1, false, false);
        }

        // Epilogue: branch-free per-lane online updates (no cross-lane ops here).
        #pragma unroll
        for (int s = 0; s < 2; ++s) {
            int gc = ct + s * 16 + l16;
            int cl = lab[gc];
            #pragma unroll
            for (int i = 0; i < 8; ++i) {
                float val = (s == 0 ? acc0[i] : acc1[i]) * TEMP_INV;
                bool diag = (gc == rowGr[i]);
                float m0 = runM[i];
                float nm = fmaxf(m0, val);                 // max INCLUDES diagonal
                float sc = __expf(m0 - nm);
                float e  = diag ? 0.f : __expf(val - nm);  // sum EXCLUDES diagonal
                runS[i] = runS[i] * sc + e;
                runM[i] = nm;
                bool pos = (!diag) && (cl == rowLab[i]);
                posS[i] += pos ? val : 0.f;
                posC[i] += pos ? 1.f : 0.f;
            }
        }
        // Protect the buffer this tile read from being overwritten at t+2.
        __syncthreads();
    }

    // ---- Final cross-lane merge within each 16-lane group (butterfly) ----
    #pragma unroll
    for (int i = 0; i < 8; ++i) {
        float m = runM[i];
        float M = m;
        #pragma unroll
        for (int x = 1; x < 16; x <<= 1) M = fmaxf(M, __shfl_xor(M, x, 32));
        float S = runS[i] * __expf(m - M);
        float P = posS[i];
        float C = posC[i];
        #pragma unroll
        for (int x = 1; x < 16; x <<= 1) {
            S += __shfl_xor(S, x, 32);
            P += __shfl_xor(P, x, 32);
            C += __shfl_xor(C, x, 32);
        }
        if (l16 == 0) {
            float logden = __logf(S + EPSV);
            float mlp = (P - C * M - C * logden) / (C + EPSV);
            float vld = (C > 0.f) ? 1.f : 0.f;
            contrib[rowGr[i]]  = mlp * vld;
            validOut[rowGr[i]] = vld;
        }
    }
}

// ---------------- Kernel 3: deterministic final reduction ----------------
__global__ __launch_bounds__(256)
void supcon_reduce(const float* __restrict__ contrib, const float* __restrict__ valid,
                   float* __restrict__ out, int B) {
    __shared__ float sC[256], sV[256];
    int t = threadIdx.x;
    float c = 0.f, v = 0.f;
    for (int i = t; i < B; i += 256) { c += contrib[i]; v += valid[i]; }
    sC[t] = c; sV[t] = v;
    __syncthreads();
    for (int s = 128; s > 0; s >>= 1) {
        if (t < s) { sC[t] += sC[t + s]; sV[t] += sV[t + s]; }
        __syncthreads();
    }
    if (t == 0) out[0] = -sC[0] / fmaxf(sV[0], 1.f);
}

extern "C" void kernel_launch(void* const* d_in, const int* in_sizes, int n_in,
                              void* d_out, int out_size, void* d_ws, size_t ws_size,
                              hipStream_t stream) {
    const float* feat = (const float*)d_in[0];
    const int*   lab  = (const int*)d_in[1];
    int B  = in_sizes[1];           // 8192
    int Dv = in_sizes[0] / B;       // 512

    _Float16* h      = (_Float16*)d_ws;                               // 8 MB f16 matrix
    float*  contrib  = (float*)((char*)d_ws + (size_t)B * Dv * 2);    // 32 KB
    float*  valid    = contrib + B;                                   // 32 KB

    normalize_f16<<<B / 8, 256, 0, stream>>>(feat, h, Dv);
    supcon_main <<<B / 128, 256, 0, stream>>>(h, lab, contrib, valid, B);
    supcon_reduce<<<1, 256, 0, stream>>>(contrib, valid, (float*)d_out, B);
}
